// GaussianBasis_22119081575300
// MI455X (gfx1250) — matrix operand order repaired and checked by reading, not compiled, not run on hardware
//
#include <hip/hip_runtime.h>
#include <hip/hip_bf16.h>

typedef __attribute__((ext_vector_type(16))) _Float16 v16h;
typedef __attribute__((ext_vector_type(8)))  float    v8f;

#define NGAUSS 2048
#define HH 256
#define WW 256
#define NPIX (HH*WW)             // 65536
#define KCHUNK 32
#define NCHUNKS (NGAUSS/KCHUNK)  // 64

// ---------------------------------------------------------------------------
// Kernel 1: per-gaussian params. Folds 0.5 and -log2(e) into the conic, and
// log2(opacity) into the FMA-chain addend, so the hot loop per weight is:
//   s = fma(ha,dx2, fma(hc,dy2, fma(hb,dxdy, lop))) ; g = exp2(s)
// i.e. 2 subs + 3 muls + 3 fmas + 1 v_exp_f32, no post-exp multiply.
// ---------------------------------------------------------------------------
__global__ void prep_params(const float* __restrict__ xyz_raw,
                            const float* __restrict__ chol_raw,
                            const float* __restrict__ opacity,
                            float4* __restrict__ P0, float2* __restrict__ P1) {
    int n = blockIdx.x * blockDim.x + threadIdx.x;
    if (n >= NGAUSS) return;
    float tx = tanhf(xyz_raw[2*n+0]);
    float ty = tanhf(xyz_raw[2*n+1]);
    float cx = 0.5f * (tx + 1.0f) * (float)WW;
    float cy = 0.5f * (ty + 1.0f) * (float)HH;
    float l1 = chol_raw[3*n+0] + 0.5f;
    float l2 = chol_raw[3*n+1];
    float l3 = chol_raw[3*n+2] + 0.5f;
    float a = l1*l1, b = l1*l2, c = l2*l2 + l3*l3;
    float inv = 1.0f / (a*c - b*b);
    float ca = c*inv, cb = -b*inv, cc = a*inv;
    const float L2E = 1.4426950408889634f;
    float op  = opacity[n];
    float lop = __log2f(op);               // exp2(s)*op == exp2(s + log2 op)
    P0[n] = make_float4(cx, cy, -0.5f*ca*L2E, -0.5f*cc*L2E);
    P1[n] = make_float2(-cb*L2E, lop);
}

// ---------------------------------------------------------------------------
// Kernel 2: pre-pack features (C,N,3) f32 -> f16 A-fragments in the exact
// 16-bit A-matrix VGPR layout (ISA 7.12.2), one contiguous v16h per lane:
//   packedF[((chunk*3 + tile)*32 + lane)*16 + j]
//   lane: m = lane&15, hi = lane>>4
//   j<8 : K = j + (hi?8:0) ; j>=8 : K = j + (hi?16:8)
// ---------------------------------------------------------------------------
__global__ void pack_features(const float* __restrict__ features,
                              _Float16* __restrict__ packedF) {
    int tid = blockIdx.x * blockDim.x + threadIdx.x;   // 0 .. 98303
    int j     =  tid        & 15;
    int lane  = (tid >> 4)  & 31;
    int tt    = (tid >> 9)  % 3;
    int chunk =  tid / (16*32*3);
    int m  = lane & 15;
    int hi = lane >> 4;
    int k  = j + ((j < 8) ? (hi ? 8 : 0) : (hi ? 16 : 8));
    int n  = chunk * KCHUNK + k;
    int cd = tt * 16 + m;
    int ch = cd / 3, d = cd - 3*ch;
    packedF[tid] = (_Float16)features[(ch * NGAUSS + n) * 3 + d];
}

// ---------------------------------------------------------------------------
// Kernel 3: fused rasterize + contraction.
// Wave = one 16-pixel tile x all 48 output rows (M = 3 WMMA tiles).
// B fragment (K=32 x N=16, f16): lane = pixel column (lane&15), K-group by
// lane>>4; each lane evaluates 16 gaussian weights per chunk from LDS params
// (16 lanes hit identical LDS addresses -> conflict-free broadcast).
// ---------------------------------------------------------------------------
__global__ __launch_bounds__(256)
void rasterize_wmma(const float4* __restrict__ P0,
                    const float2* __restrict__ P1,
                    const _Float16* __restrict__ packedF,
                    float* __restrict__ out) {
    __shared__ float4 sP0[NGAUSS];   // 32 KB
    __shared__ float2 sP1[NGAUSS];   // 16 KB

    int t = threadIdx.x;
    for (int i = t; i < NGAUSS; i += 256) { sP0[i] = P0[i]; sP1[i] = P1[i]; }
    __syncthreads();

    int lane = t & 31;
    int warp = t >> 5;
    int tile = blockIdx.x * 8 + warp;      // 512 blocks * 8 waves = 4096 tiles
    int pixbase = tile * 16;
    int col = lane & 15;
    int hi  = lane >> 4;
    int p   = pixbase + col;
    float px = (float)(p & (WW - 1)) + 0.5f;
    float py = (float)(p >> 8) + 0.5f;
    int kgrp = hi * 8;

    v8f acc0 = {}, acc1 = {}, acc2 = {};
    const v16h* __restrict__ fptr = (const v16h*)packedF;

    for (int chunk = 0; chunk < NCHUNKS; ++chunk) {
        // ---- build B fragment: 16 weights for gaussians nbase+[0,8) and nbase+[16,24)
        v16h b;
        int nbase = chunk * KCHUNK + kgrp;
#pragma unroll
        for (int j = 0; j < 16; ++j) {
            float4 q = sP0[nbase + j + (j >= 8 ? 8 : 0)];       // cx, cy, ha, hc  (LDS broadcast)
            float2 r = sP1[nbase + j + (j >= 8 ? 8 : 0)];       // hb, log2(opacity)
            float dx = px - q.x;
            float dy = py - q.y;
            float s  = fmaf(q.z, dx * dx,
                       fmaf(q.w, dy * dy,
                       fmaf(r.x, dx * dy, r.y)));
            b[j] = (_Float16)__builtin_amdgcn_exp2f(s);
        }
        // ---- A fragments: contiguous 32B per lane, L2-resident
        int abase = (chunk * 3) * 32 + lane;
        v16h a0 = fptr[abase];
        v16h a1 = fptr[abase + 32];
        v16h a2 = fptr[abase + 64];
        // ---- D = A*B + C  (one B reused across all 48 output rows)
        acc0 = __builtin_amdgcn_wmma_f32_16x16x32_f16(false, a0, false, b,
                                                      (short)0, acc0, false, false);
        acc1 = __builtin_amdgcn_wmma_f32_16x16x32_f16(false, a1, false, b,
                                                      (short)0, acc1, false, false);
        acc2 = __builtin_amdgcn_wmma_f32_16x16x32_f16(false, a2, false, b,
                                                      (short)0, acc2, false, false);
    }

    // C/D layout: VGPR r, lanes 0-15 -> M=r, lanes 16-31 -> M=r+8; col = lane&15
    int base = pixbase + col + hi * 8 * NPIX;
#pragma unroll
    for (int r = 0; r < 8; ++r) {
        out[r * NPIX + base             ] = acc0[r];   // cd tile 0
        out[r * NPIX + base + 16 * NPIX ] = acc1[r];   // cd tile 1
        out[r * NPIX + base + 32 * NPIX ] = acc2[r];   // cd tile 2
    }
}

// ---------------------------------------------------------------------------
extern "C" void kernel_launch(void* const* d_in, const int* in_sizes, int n_in,
                              void* d_out, int out_size, void* d_ws, size_t ws_size,
                              hipStream_t stream) {
    const float* xyz  = (const float*)d_in[0];   // (N,2)
    const float* chol = (const float*)d_in[1];   // (N,3)
    const float* feat = (const float*)d_in[2];   // (C,N,3)
    const float* opac = (const float*)d_in[3];   // (N,1)
    float* out = (float*)d_out;                  // (C,3,H,W) f32

    char* ws = (char*)d_ws;
    float4*   P0      = (float4*)(ws);            // 32 KB
    float2*   P1      = (float2*)(ws + 32768);    // 16 KB
    _Float16* packedF = (_Float16*)(ws + 49152);  // 192 KB

    prep_params  <<<  8, 256, 0, stream>>>(xyz, chol, opac, P0, P1);
    pack_features<<<384, 256, 0, stream>>>(feat, packedF);
    rasterize_wmma<<<512, 256, 0, stream>>>(P0, P1, packedF, out);
}
